// SelfAttention_46729244180658
// MI455X (gfx1250) — matrix operand order repaired and checked by reading, not compiled, not run on hardware
//
#include <hip/hip_runtime.h>

// ---------------------------------------------------------------------------
// Causal MHA for MI455X (gfx1250, wave32, WMMA 16x16x32 bf16 / f32 accum)
//   B=4, T=2048, C=1024, H=16, hd=64
// ---------------------------------------------------------------------------

#define B_ 4
#define T_ 2048
#define C_ 1024
#define H_ 16
#define HD 64

typedef __attribute__((ext_vector_type(16))) __bf16 v16bf;
typedef __attribute__((ext_vector_type(8)))  __bf16 v8bf;
typedef __attribute__((ext_vector_type(8)))  float  v8f;
typedef __attribute__((ext_vector_type(4)))  float  v4f;

union V16 { v16bf v; v8bf h[2]; };

__device__ __forceinline__ v8f wmma_bf16(v16bf a, v16bf b, v8f c) {
  return __builtin_amdgcn_wmma_f32_16x16x32_bf16(
      false, a, false, b, (short)0, c, false, false);
}

// Async global->LDS copy, 16 bytes per lane (tracked with ASYNCcnt).
__device__ __forceinline__ void async_b128(const void* g, const void* l) {
  unsigned lo = (unsigned)(uintptr_t)l;   // low 32 bits of generic = LDS offset
  asm volatile("global_load_async_to_lds_b128 %0, %1, off"
               :: "v"(lo), "v"(g) : "memory");
}
__device__ __forceinline__ void wait_async0() {
  asm volatile("s_wait_asynccnt 0x0" ::: "memory");
}

// A fragment (16x32) from f32 row-major source, converted to bf16
__device__ __forceinline__ void load_a_f32(V16& a, const float* xrow, int k0, int lh) {
  const float* p0 = xrow + k0 + 16 * lh;
  const float* p1 = xrow + k0 + 16 * lh + 8;
  v4f f0 = *(const v4f*)(p0);
  v4f f1 = *(const v4f*)(p0 + 4);
  v4f f2 = *(const v4f*)(p1);
  v4f f3 = *(const v4f*)(p1 + 4);
#pragma unroll
  for (int e = 0; e < 4; ++e) {
    a.v[e]      = (__bf16)f0[e];
    a.v[e + 4]  = (__bf16)f1[e];
    a.v[e + 8]  = (__bf16)f2[e];
    a.v[e + 12] = (__bf16)f3[e];
  }
}

__device__ __forceinline__ void load_a_bf16(V16& a, const __bf16* row, int k0, int lh) {
  a.h[0] = *(const v8bf*)(row + k0 + 16 * lh);
  a.h[1] = *(const v8bf*)(row + k0 + 16 * lh + 8);
}

// 4 B fragments (32x16 each) from bf16 [N][K] source (K contiguous)
__device__ __forceinline__ void load_b4(V16 (&b)[4], const __bf16* Wt,
                                        int n0, int ln, int lh, int k0) {
#pragma unroll
  for (int c = 0; c < 4; ++c) {
    const __bf16* bp = Wt + (size_t)(n0 + c * 16 + ln) * C_ + k0 + 16 * lh;
    b[c].h[0] = *(const v8bf*)(bp);
    b[c].h[1] = *(const v8bf*)(bp + 8);
  }
}

// ---------------------------------------------------------------------------
// Transpose-convert f32 [K][N] -> bf16 [N][K]
// ---------------------------------------------------------------------------
__global__ void __launch_bounds__(256)
k_transpose_bf16(const float* __restrict__ in, __bf16* __restrict__ out,
                 int K, int N) {
  int idx = blockIdx.x * 256 + threadIdx.x;
  if (idx >= K * N) return;
  int n = idx / K;
  int k = idx - n * K;
  out[idx] = (__bf16)in[(size_t)k * N + n];
}

// ---------------------------------------------------------------------------
// QKV GEMM: qkv = x[8192,1024] @ Wqkv[1024,3072] + b, double-buffered.
//   Epilogue: uniform head/stream select; Q,K row-major, V transposed.
// ---------------------------------------------------------------------------
__global__ void __launch_bounds__(256)
k_qkv_gemm(const float* __restrict__ X, const __bf16* __restrict__ Wt,
           const float* __restrict__ bias,
           __bf16* __restrict__ Qb, __bf16* __restrict__ Kb,
           __bf16* __restrict__ Vt) {
  const int wave = blockIdx.x * 8 + (threadIdx.x >> 5);
  const int lane = threadIdx.x & 31;
  const int lh = lane >> 4, ln = lane & 15;
  const int NG = (3 * C_) / 64;                 // 48
  const int mt = wave / NG;
  const int ng = wave - mt * NG;
  const int m0 = mt * 16, n0 = ng * 64;

  v8f acc[4];
#pragma unroll
  for (int c = 0; c < 4; ++c) acc[c] = (v8f)0.f;

  const float* xrow = X + (size_t)(m0 + ln) * C_;

  // ping-pong double buffer: prefetch k+32 while WMMAing k
  V16 a0, a1, b0[4], b1[4];
  load_a_f32(a0, xrow, 0, lh);
  load_b4(b0, Wt, n0, ln, lh, 0);
  for (int k0 = 0; k0 < C_; k0 += 64) {
    load_a_f32(a1, xrow, k0 + 32, lh);
    load_b4(b1, Wt, n0, ln, lh, k0 + 32);
#pragma unroll
    for (int c = 0; c < 4; ++c) acc[c] = wmma_bf16(a0.v, b0[c].v, acc[c]);
    if (k0 + 64 < C_) {
      load_a_f32(a0, xrow, k0 + 64, lh);
      load_b4(b0, Wt, n0, ln, lh, k0 + 64);
    }
#pragma unroll
    for (int c = 0; c < 4; ++c) acc[c] = wmma_bf16(a1.v, b1[c].v, acc[c]);
  }

  // ---- epilogue: s (q/k/v) and h are wave-uniform (n0 is 64-aligned) ----
  const int s = n0 >> 10;
  const int h = (n0 & (C_ - 1)) >> 6;
  const int bq = m0 >> 11;
  const int tb = m0 & (T_ - 1);                 // tile base t (rows tb..tb+15)
  const size_t bh = (size_t)(bq * H_ + h);

  if (s < 2) {
    __bf16* dst = (s == 0 ? Qb : Kb) + (bh * T_ + tb) * HD;
#pragma unroll
    for (int c = 0; c < 4; ++c) {
      const int d = c * 16 + ln;
      const float bv = bias[n0 + d];
#pragma unroll
      for (int r = 0; r < 8; ++r)
        dst[(size_t)(r + 8 * lh) * HD + d] = (__bf16)(acc[c][r] + bv);
    }
  } else {
    // V^T: per lane, 8 consecutive t -> one packed b128 store per tile
    __bf16* dst = Vt + bh * HD * T_ + tb + 8 * lh;
#pragma unroll
    for (int c = 0; c < 4; ++c) {
      const int d = c * 16 + ln;
      const float bv = bias[n0 + d];
      v8bf pk;
#pragma unroll
      for (int r = 0; r < 8; ++r) pk[r] = (__bf16)(acc[c][r] + bv);
      *(v8bf*)(dst + (size_t)d * T_) = pk;
    }
  }
}

// ---------------------------------------------------------------------------
// Flash attention, block-cooperative K/V staging via async-to-LDS.
//   Block: 8 waves, same bh, q-tiles qg*8..qg*8+7. Uniform key loop.
// ---------------------------------------------------------------------------
__global__ void __launch_bounds__(256)
k_attn(const __bf16* __restrict__ Qb, const __bf16* __restrict__ Kb,
       const __bf16* __restrict__ Vt, __bf16* __restrict__ Y) {
  __shared__ __align__(16) __bf16 ldsK[32 * 64];      // [key][d]   4KB
  __shared__ __align__(16) __bf16 ldsV[64 * 32];      // [d][key]   4KB
  __shared__ __align__(16) __bf16 ldsP[8][16 * 32];   // per-wave P 8KB
  const int tid  = threadIdx.x;
  const int wib  = tid >> 5;
  const int lane = tid & 31;
  const int lh = lane >> 4, ln = lane & 15;
  const int bh = blockIdx.x >> 4;               // 0..63
  const int qg = blockIdx.x & 15;               // 0..15
  const int qt = qg * 8 + wib;
  const int t0 = qt * 16;
  __bf16* pls = &ldsP[wib][0];

  // Q fragments (A-layout), loaded once per wave
  const __bf16* qrow = Qb + ((size_t)bh * T_ + t0 + ln) * HD;
  V16 qa0, qa1;
  qa0.h[0] = *(const v8bf*)(qrow + 16 * lh);
  qa0.h[1] = *(const v8bf*)(qrow + 16 * lh + 8);
  qa1.h[0] = *(const v8bf*)(qrow + 32 + 16 * lh);
  qa1.h[1] = *(const v8bf*)(qrow + 40 + 16 * lh);

  v8f o0 = (v8f)0.f, o1 = (v8f)0.f, o2 = (v8f)0.f, o3 = (v8f)0.f;
  float mst[8], lst[8];
#pragma unroll
  for (int r = 0; r < 8; ++r) { mst[r] = -3.0e38f; lst[r] = 0.f; }

  const float scale = 0.125f;
  const int kbmax = qg * 4 + 3;                 // block-uniform (causal max)

  // cooperative staging indices (256 threads, 16B each)
  const int krw = tid >> 3, krc = tid & 7;      // K: 32 rows x 8 chunks
  const int vrw = tid >> 2, vrc = tid & 3;      // V: 64 rows x 4 chunks

  for (int kb = 0; kb <= kbmax; ++kb) {
    const int kbase = kb * 32;
    // ---- async stage K(32x64) and V^T(64x32) tiles into LDS ----
    async_b128(Kb + ((size_t)bh * T_ + kbase + krw) * HD + krc * 8,
               &ldsK[krw * 64 + krc * 8]);
    async_b128(Vt + ((size_t)bh * HD + vrw) * T_ + kbase + vrc * 8,
               &ldsV[vrw * 32 + vrc * 8]);
    wait_async0();
    __syncthreads();

    // ---- S = Q . K^T from LDS (B fragment: key col, d contiguous) ----
    v8f s0 = (v8f)0.f, s1 = (v8f)0.f;
    {
      const __bf16* kr = &ldsK[ln * 64 + 16 * lh];
      V16 f0, f1;
      f0.h[0] = *(const v8bf*)(kr);       f0.h[1] = *(const v8bf*)(kr + 8);
      f1.h[0] = *(const v8bf*)(kr + 32);  f1.h[1] = *(const v8bf*)(kr + 40);
      s0 = wmma_bf16(qa0.v, f0.v, s0);
      s0 = wmma_bf16(qa1.v, f1.v, s0);
      const __bf16* kr2 = kr + 16 * 64;
      f0.h[0] = *(const v8bf*)(kr2);      f0.h[1] = *(const v8bf*)(kr2 + 8);
      f1.h[0] = *(const v8bf*)(kr2 + 32); f1.h[1] = *(const v8bf*)(kr2 + 40);
      s1 = wmma_bf16(qa0.v, f0.v, s1);
      s1 = wmma_bf16(qa1.v, f1.v, s1);
    }
    // ---- scale + causal mask ----
#pragma unroll
    for (int r = 0; r < 8; ++r) {
      const int qr = t0 + r + 8 * lh;
      s0[r] = (kbase + ln      <= qr) ? s0[r] * scale : -3.0e38f;
      s1[r] = (kbase + 16 + ln <= qr) ? s1[r] * scale : -3.0e38f;
    }
    // ---- online softmax ----
#pragma unroll
    for (int r = 0; r < 8; ++r) {
      float rm = fmaxf(s0[r], s1[r]);
      rm = fmaxf(rm, __shfl_xor(rm, 1, 32));
      rm = fmaxf(rm, __shfl_xor(rm, 2, 32));
      rm = fmaxf(rm, __shfl_xor(rm, 4, 32));
      rm = fmaxf(rm, __shfl_xor(rm, 8, 32));
      const float mnew = fmaxf(mst[r], rm);
      const float p0 = __expf(s0[r] - mnew);
      const float p1 = __expf(s1[r] - mnew);
      float rs = p0 + p1;
      rs += __shfl_xor(rs, 1, 32);
      rs += __shfl_xor(rs, 2, 32);
      rs += __shfl_xor(rs, 4, 32);
      rs += __shfl_xor(rs, 8, 32);
      const float alpha = __expf(mst[r] - mnew);
      lst[r] = lst[r] * alpha + rs;
      mst[r] = mnew;
      o0[r] *= alpha; o1[r] *= alpha; o2[r] *= alpha; o3[r] *= alpha;
      const int row = r + 8 * lh;
      pls[row * 32 + ln]      = (__bf16)p0;
      pls[row * 32 + 16 + ln] = (__bf16)p1;
    }
    __builtin_amdgcn_wave_barrier();
    V16 pA;
    pA.h[0] = *(const v8bf*)(pls + ln * 32 + 16 * lh);
    pA.h[1] = *(const v8bf*)(pls + ln * 32 + 16 * lh + 8);
    __builtin_amdgcn_wave_barrier();
    // ---- O += P . V from LDS ----
    {
      V16 vB;
      const __bf16* vp0 = &ldsV[(ln)      * 32 + 16 * lh];
      vB.h[0] = *(const v8bf*)(vp0); vB.h[1] = *(const v8bf*)(vp0 + 8);
      o0 = wmma_bf16(pA.v, vB.v, o0);
      const __bf16* vp1 = &ldsV[(16 + ln) * 32 + 16 * lh];
      vB.h[0] = *(const v8bf*)(vp1); vB.h[1] = *(const v8bf*)(vp1 + 8);
      o1 = wmma_bf16(pA.v, vB.v, o1);
      const __bf16* vp2 = &ldsV[(32 + ln) * 32 + 16 * lh];
      vB.h[0] = *(const v8bf*)(vp2); vB.h[1] = *(const v8bf*)(vp2 + 8);
      o2 = wmma_bf16(pA.v, vB.v, o2);
      const __bf16* vp3 = &ldsV[(48 + ln) * 32 + 16 * lh];
      vB.h[0] = *(const v8bf*)(vp3); vB.h[1] = *(const v8bf*)(vp3 + 8);
      o3 = wmma_bf16(pA.v, vB.v, o3);
    }
    __syncthreads();   // protect ldsK/ldsV before next stage
  }

  // ---- finalize + store y (bf16, [B*T][C]) ----
  const int b = bh >> 4, h = bh & 15;
#pragma unroll
  for (int r = 0; r < 8; ++r) {
    const float inv = 1.0f / lst[r];
    const size_t row = (size_t)(b * T_ + t0 + r + 8 * lh) * C_ + h * HD;
    Y[row + ln]      = (__bf16)(o0[r] * inv);
    Y[row + 16 + ln] = (__bf16)(o1[r] * inv);
    Y[row + 32 + ln] = (__bf16)(o2[r] * inv);
    Y[row + 48 + ln] = (__bf16)(o3[r] * inv);
  }
}

// ---------------------------------------------------------------------------
// Proj GEMM: out = y[8192,1024] @ Wproj[1024,1024] + b (f32), double-buffered
// ---------------------------------------------------------------------------
__global__ void __launch_bounds__(256)
k_proj_gemm(const __bf16* __restrict__ Yb, const __bf16* __restrict__ Wt,
            const float* __restrict__ bias, float* __restrict__ out) {
  const int wave = blockIdx.x * 8 + (threadIdx.x >> 5);
  const int lane = threadIdx.x & 31;
  const int lh = lane >> 4, ln = lane & 15;
  const int NG = C_ / 64;                       // 16
  const int mt = wave / NG;
  const int ng = wave - mt * NG;
  const int m0 = mt * 16, n0 = ng * 64;

  v8f acc[4];
#pragma unroll
  for (int c = 0; c < 4; ++c) acc[c] = (v8f)0.f;

  const __bf16* yrow = Yb + (size_t)(m0 + ln) * C_;

  V16 a0, a1, b0[4], b1[4];
  load_a_bf16(a0, yrow, 0, lh);
  load_b4(b0, Wt, n0, ln, lh, 0);
  for (int k0 = 0; k0 < C_; k0 += 64) {
    load_a_bf16(a1, yrow, k0 + 32, lh);
    load_b4(b1, Wt, n0, ln, lh, k0 + 32);
#pragma unroll
    for (int c = 0; c < 4; ++c) acc[c] = wmma_bf16(a0.v, b0[c].v, acc[c]);
    if (k0 + 64 < C_) {
      load_a_bf16(a0, yrow, k0 + 64, lh);
      load_b4(b0, Wt, n0, ln, lh, k0 + 64);
    }
#pragma unroll
    for (int c = 0; c < 4; ++c) acc[c] = wmma_bf16(a1.v, b1[c].v, acc[c]);
  }
#pragma unroll
  for (int c = 0; c < 4; ++c) {
    const int n = n0 + c * 16 + ln;
    const float bv = bias[n];
#pragma unroll
    for (int r = 0; r < 8; ++r) {
      const int m = m0 + r + 8 * lh;
      out[(size_t)m * C_ + n] = acc[c][r] + bv;
    }
  }
}

// ---------------------------------------------------------------------------
extern "C" void kernel_launch(void* const* d_in, const int* in_sizes, int n_in,
                              void* d_out, int out_size, void* d_ws, size_t ws_size,
                              hipStream_t stream) {
  const float* x      = (const float*)d_in[0];
  const float* W_qkv  = (const float*)d_in[1];
  const float* b_qkv  = (const float*)d_in[2];
  const float* W_proj = (const float*)d_in[3];
  const float* b_proj = (const float*)d_in[4];
  float* out = (float*)d_out;

  uint8_t* ws = (uint8_t*)d_ws;
  size_t off = 0;
  auto alloc = [&](size_t bytes) {
    void* p = ws + off;
    off = (off + bytes + 255) & ~(size_t)255;
    return p;
  };
  __bf16* WqkvT  = (__bf16*)alloc((size_t)3 * C_ * C_ * 2);   // [3072][1024]
  __bf16* WprojT = (__bf16*)alloc((size_t)C_ * C_ * 2);       // [1024][1024]
  __bf16* Qb     = (__bf16*)alloc((size_t)B_ * T_ * C_ * 2);  // [bh][t][d]
  __bf16* Kb     = (__bf16*)alloc((size_t)B_ * T_ * C_ * 2);  // [bh][t][d]
  __bf16* Vt     = (__bf16*)alloc((size_t)B_ * T_ * C_ * 2);  // [bh][d][t]
  __bf16* Yb     = (__bf16*)alloc((size_t)B_ * T_ * C_ * 2);  // [B*T][C]
  (void)ws_size; (void)in_sizes; (void)n_in; (void)out_size;

  k_transpose_bf16<<<(3 * C_ * C_) / 256, 256, 0, stream>>>(W_qkv, WqkvT, C_, 3 * C_);
  k_transpose_bf16<<<(C_ * C_) / 256, 256, 0, stream>>>(W_proj, WprojT, C_, C_);
  k_qkv_gemm<<<3072, 256, 0, stream>>>(x, WqkvT, b_qkv, Qb, Kb, Vt);
  k_attn<<<1024, 256, 0, stream>>>(Qb, Kb, Vt, Yb);
  k_proj_gemm<<<1024, 256, 0, stream>>>(Yb, WprojT, b_proj, out);
}
